// SelfAttention_19524921327924
// MI455X (gfx1250) — matrix operand order repaired and checked by reading, not compiled, not run on hardware
//
#include <hip/hip_runtime.h>

// ---------------------------------------------------------------------------
// Types
// ---------------------------------------------------------------------------
typedef __bf16 bf16;
typedef bf16  v16bf __attribute__((ext_vector_type(16)));
typedef bf16  v8bf  __attribute__((ext_vector_type(8)));
typedef float v8f   __attribute__((ext_vector_type(8)));

// GCC-style vector to match the async-to-LDS builtin prototype (V4i).
typedef int gv4i __attribute__((vector_size(16)));
typedef __attribute__((address_space(1))) gv4i* g4p;
typedef __attribute__((address_space(3))) gv4i* l4p;

#ifndef __has_builtin
#define __has_builtin(x) 0
#endif

#if __has_builtin(__builtin_amdgcn_global_load_async_to_lds_b128)
#define HAS_ASYNC 1
#else
#define HAS_ASYNC 0
#endif

// 16-byte global->LDS copy. Async (ASYNCcnt-tracked, no VGPR round trip) when
// the gfx1250 builtin is available, otherwise plain load+store fallback.
static __device__ __forceinline__ void cp16(const bf16* g, bf16* l) {
#if HAS_ASYNC
  // Generic global address == AS1 address; generic LDS address keeps the LDS
  // byte offset in addr[31:0] (ISA aperture rules), so integer casts are exact.
  __builtin_amdgcn_global_load_async_to_lds_b128(
      (g4p)(unsigned long long)g,
      (l4p)(unsigned int)(unsigned long long)l,
      0, 0);
#else
  *(v8bf*)l = *(const v8bf*)g;
#endif
}

static __device__ __forceinline__ void wait_async() {
#if HAS_ASYNC
#if __has_builtin(__builtin_amdgcn_s_wait_asynccnt)
  __builtin_amdgcn_s_wait_asynccnt(0);
#else
  asm volatile("s_wait_asynccnt 0x0" ::: "memory");
#endif
#endif
}

static __device__ __forceinline__ void wait_lds() {
  asm volatile("s_wait_dscnt 0x0" ::: "memory");
}

static __device__ __forceinline__ bf16 f2b(float f) {
  unsigned u = __builtin_bit_cast(unsigned, f);
  unsigned r = (u + 0x7FFFu + ((u >> 16) & 1u)) >> 16;
  unsigned short us = (unsigned short)r;
  return __builtin_bit_cast(bf16, us);
}

static __device__ __forceinline__ v16bf make16(v8bf lo, v8bf hi) {
  return __builtin_shufflevector(lo, hi, 0,1,2,3,4,5,6,7,8,9,10,11,12,13,14,15);
}

// ---------------------------------------------------------------------------
// Problem constants (B=4, T=2048, C=1024, H=16, hd=64)
// ---------------------------------------------------------------------------
#define TB 4
#define TT 2048
#define TC 1024
#define TH 16
#define TE 64

// ---------------------------------------------------------------------------
// fp32 -> bf16 elementwise conversion
// ---------------------------------------------------------------------------
__global__ void f2bf_kernel(const float* __restrict__ in, bf16* __restrict__ out,
                            size_t n) {
  size_t i = (size_t)blockIdx.x * blockDim.x + threadIdx.x;
  size_t stride = (size_t)gridDim.x * blockDim.x;
  for (; i < n; i += stride) out[i] = f2b(in[i]);
}

// ---------------------------------------------------------------------------
// GEMM: out = A(bf16, MxK row-major) @ W(bf16, NxK row-major)^T + bias
// mode 0: store fp32 to outF[M*N]
// mode 1: scatter bf16 into qkv buffers [which][B,H,T,64], q scaled by 0.125
// Block: 256 threads (8 waves), tile 128(M) x 128(N), K-step 32,
// double-buffered LDS with async global->LDS copies.
// Wave (wm = wave>>1, wn = wave&1) owns a 32(M) x 64(N) output strip.
// ---------------------------------------------------------------------------
__global__ __launch_bounds__(256) void gemm_xwT(
    const bf16* __restrict__ A, const bf16* __restrict__ W,
    const float* __restrict__ bias, int M, int N, int K, int mode,
    float* __restrict__ outF, bf16* __restrict__ qkv) {
  __shared__ __align__(16) bf16 sA[2][128 * 32];
  __shared__ __align__(16) bf16 sW[2][128 * 32];

  const int tid  = threadIdx.x;        // 0..255
  const int lane = tid & 31;
  const int wave = tid >> 5;           // 0..7
  const int wm   = wave >> 1;          // 0..3 : 32-row strip
  const int wn   = wave & 1;           // 0..1 : 64-col strip
  const int hi   = (lane >> 4) & 1;
  const int ln   = lane & 15;
  const int m0   = blockIdx.y * 128;
  const int n0   = blockIdx.x * 128;

  // Tile loads: 128x32 halves = 512 16B-chunks per tile; 2 chunks per thread.
  const int r0 = tid >> 2;             // 0..63
  const int c0 = (tid & 3) * 8;        // 0,8,16,24

  v8f acc[2][4] = {};
  const int nk = K >> 5;

  // Prologue prefetch into buffer 0
  {
    cp16(&A[(size_t)(m0 + r0) * K + c0],      &sA[0][r0 * 32 + c0]);
    cp16(&A[(size_t)(m0 + r0 + 64) * K + c0], &sA[0][(r0 + 64) * 32 + c0]);
    cp16(&W[(size_t)(n0 + r0) * K + c0],      &sW[0][r0 * 32 + c0]);
    cp16(&W[(size_t)(n0 + r0 + 64) * K + c0], &sW[0][(r0 + 64) * 32 + c0]);
  }

  for (int it = 0; it < nk; ++it) {
    wait_async();
    __syncthreads();
    if (it + 1 < nk) {
      const int kk = (it + 1) << 5;
      const int nb = (it + 1) & 1;
      cp16(&A[(size_t)(m0 + r0) * K + kk + c0],      &sA[nb][r0 * 32 + c0]);
      cp16(&A[(size_t)(m0 + r0 + 64) * K + kk + c0], &sA[nb][(r0 + 64) * 32 + c0]);
      cp16(&W[(size_t)(n0 + r0) * K + kk + c0],      &sW[nb][r0 * 32 + c0]);
      cp16(&W[(size_t)(n0 + r0 + 64) * K + kk + c0], &sW[nb][(r0 + 64) * 32 + c0]);
    }
    const bf16* a_ = sA[it & 1];
    const bf16* w_ = sW[it & 1];
    const int aoff = hi ? 8 : 0;
    const int koff = hi ? 8 : 0;

    v16bf af[2];
    #pragma unroll
    for (int mt = 0; mt < 2; ++mt) {
      const bf16* ap = &a_[(wm * 32 + mt * 16 + ln) * 32];
      af[mt] = make16(*(const v8bf*)(ap + aoff), *(const v8bf*)(ap + 16 + aoff));
    }
    #pragma unroll
    for (int nt = 0; nt < 4; ++nt) {
      const bf16* bp = &w_[(wn * 64 + nt * 16 + ln) * 32 + koff];
      v16bf bfv = make16(*(const v8bf*)bp, *(const v8bf*)(bp + 16));
      #pragma unroll
      for (int mt = 0; mt < 2; ++mt)
        acc[mt][nt] = __builtin_amdgcn_wmma_f32_16x16x32_bf16(
            false, af[mt], false, bfv, (short)0, acc[mt][nt], false, false);
    }
  }

  // C/D layout: lane owns column n=ln, rows m = r + 8*hi
  #pragma unroll
  for (int mt = 0; mt < 2; ++mt) {
    #pragma unroll
    for (int nt = 0; nt < 4; ++nt) {
      #pragma unroll
      for (int r = 0; r < 8; ++r) {
        int m   = r + hi * 8;
        int row = m0 + wm * 32 + mt * 16 + m;
        int col = n0 + wn * 64 + nt * 16 + ln;
        float v = acc[mt][nt][r] + bias[col];
        if (mode == 0) {
          outF[(size_t)row * N + col] = v;
        } else {
          int which = col >> 10;      // 0=q 1=k 2=v
          int c     = col & 1023;
          int h     = c >> 6, e = c & 63;
          int b     = row >> 11, t = row & 2047;
          if (which == 0) v *= 0.125f;  // 1/sqrt(64)
          size_t per = (size_t)TB * TH * TT * TE;
          qkv[(size_t)which * per +
              ((((size_t)b * TH + h) * TT + t) * TE + e)] = f2b(v);
        }
      }
    }
  }
}

// ---------------------------------------------------------------------------
// Causal flash attention, bf16 WMMA, f32 accumulation.
// Grid: (T/64, B*H). Block: 128 threads (4 waves). Each wave: 16 query rows.
// Q pre-scaled by 1/sqrt(hd). Output -> Y bf16 [B,T,C] with C = h*64+d.
// ---------------------------------------------------------------------------
__global__ __launch_bounds__(128) void attn_kernel(
    const bf16* __restrict__ Q, const bf16* __restrict__ Kb,
    const bf16* __restrict__ Vb, bf16* __restrict__ Y) {
  __shared__ __align__(16) bf16 sQ[64 * 64];
  __shared__ __align__(16) bf16 sK[64 * 64];
  __shared__ __align__(16) bf16 sVt[64 * 64];   // transposed: [d][key]
  __shared__ __align__(16) bf16 sP[4][16 * 64]; // per-wave P tile

  const int tid  = threadIdx.x;
  const int lane = tid & 31;
  const int wave = tid >> 5;
  const int hi   = (lane >> 4) & 1;
  const int ln   = lane & 15;
  const int qt    = blockIdx.x;
  const int bh    = blockIdx.y;
  const int qbase = qt * 64;

  const bf16* Qh = Q  + (size_t)bh * TT * TE;
  const bf16* Kh = Kb + (size_t)bh * TT * TE;
  const bf16* Vh = Vb + (size_t)bh * TT * TE;

  // Load Q tile (64 rows x 64) via async copies
  #pragma unroll
  for (int i = 0; i < 4; ++i) {
    int c = tid + i * 128;            // 0..511
    int r = c >> 3, col = (c & 7) * 8;
    cp16(&Qh[(size_t)(qbase + r) * 64 + col], &sQ[r * 64 + col]);
  }
  wait_async();
  __syncthreads();

  // Q A-fragments held in registers (two K-steps of 32)
  v16bf qf[2];
  {
    const bf16* qp = &sQ[(wave * 16 + ln) * 64];
    const int aoff = hi ? 8 : 0;
    #pragma unroll
    for (int ks = 0; ks < 2; ++ks)
      qf[ks] = make16(*(const v8bf*)(qp + ks * 32 + aoff),
                      *(const v8bf*)(qp + ks * 32 + 16 + aoff));
  }

  v8f o[4] = {};
  float rmax[8], rsum[8];
  #pragma unroll
  for (int r = 0; r < 8; ++r) { rmax[r] = -3.0e38f; rsum[r] = 0.f; }

  const int koff = hi ? 8 : 0;

  for (int kt = 0; kt <= qt; ++kt) {
    const int kb = kt * 64;
    __syncthreads();
    // K tile: async row-major; V tile: sync load + transpose into sVt
    #pragma unroll
    for (int i = 0; i < 4; ++i) {
      int c = tid + i * 128;
      int r = c >> 3, col = (c & 7) * 8;
      cp16(&Kh[(size_t)(kb + r) * 64 + col], &sK[r * 64 + col]);
      v8bf vv = *(const v8bf*)&Vh[(size_t)(kb + r) * 64 + col];
      #pragma unroll
      for (int j = 0; j < 8; ++j) sVt[(col + j) * 64 + r] = vv[j];
    }
    wait_async();
    __syncthreads();

    // S = Q @ K^T  (16x64 per wave)
    v8f s[4] = {};
    #pragma unroll
    for (int nt = 0; nt < 4; ++nt) {
      #pragma unroll
      for (int ks = 0; ks < 2; ++ks) {
        const bf16* bp = &sK[(nt * 16 + ln) * 64 + ks * 32 + koff];
        v16bf bfv = make16(*(const v8bf*)bp, *(const v8bf*)(bp + 16));
        s[nt] = __builtin_amdgcn_wmma_f32_16x16x32_bf16(
            false, qf[ks], false, bfv, (short)0, s[nt], false, false);
      }
    }

    // Causal mask — only the diagonal tile needs it
    if (kt == qt) {
      #pragma unroll
      for (int nt = 0; nt < 4; ++nt) {
        int j = kb + nt * 16 + ln;
        #pragma unroll
        for (int r = 0; r < 8; ++r) {
          int irow = qbase + wave * 16 + r + hi * 8;
          if (j > irow) s[nt][r] = -3.0e38f;
        }
      }
    }

    // Online softmax (row stats broadcast across 16-lane column groups)
    float corr[8];
    #pragma unroll
    for (int r = 0; r < 8; ++r) {
      float m_ = fmaxf(fmaxf(s[0][r], s[1][r]), fmaxf(s[2][r], s[3][r]));
      #pragma unroll
      for (int msk = 1; msk < 16; msk <<= 1) m_ = fmaxf(m_, __shfl_xor(m_, msk, 32));
      float nm = fmaxf(rmax[r], m_);
      corr[r]  = __expf(rmax[r] - nm);
      rmax[r]  = nm;
      float psum = 0.f;
      #pragma unroll
      for (int nt = 0; nt < 4; ++nt) {
        float p = __expf(s[nt][r] - nm);
        s[nt][r] = p;
        psum += p;
      }
      #pragma unroll
      for (int msk = 1; msk < 16; msk <<= 1) psum += __shfl_xor(psum, msk, 32);
      rsum[r] = rsum[r] * corr[r] + psum;
      #pragma unroll
      for (int nt = 0; nt < 4; ++nt) o[nt][r] *= corr[r];
    }

    // Re-layout P: C-fragment -> LDS -> A-fragment. P tile is wave-private;
    // same-wave LDS ops are in-order, so a dscnt wait (compiler fence) suffices.
    #pragma unroll
    for (int nt = 0; nt < 4; ++nt)
      #pragma unroll
      for (int r = 0; r < 8; ++r)
        sP[wave][(r + hi * 8) * 64 + nt * 16 + ln] = f2b(s[nt][r]);
    wait_lds();

    // O += P @ V
    {
      const bf16* pp = &sP[wave][ln * 64];
      const int aoff = hi ? 8 : 0;
      #pragma unroll
      for (int ks = 0; ks < 2; ++ks) {
        v16bf pf = make16(*(const v8bf*)(pp + ks * 32 + aoff),
                          *(const v8bf*)(pp + ks * 32 + 16 + aoff));
        #pragma unroll
        for (int nt = 0; nt < 4; ++nt) {
          const bf16* bp = &sVt[(nt * 16 + ln) * 64 + ks * 32 + koff];
          v16bf bv = make16(*(const v8bf*)bp, *(const v8bf*)(bp + 16));
          o[nt] = __builtin_amdgcn_wmma_f32_16x16x32_bf16(
              false, pf, false, bv, (short)0, o[nt], false, false);
        }
      }
    }
  }

  // Normalize and store Y (bf16, [B,T,C] layout)
  const int b = bh >> 4, h = bh & 15;
  #pragma unroll
  for (int nt = 0; nt < 4; ++nt) {
    #pragma unroll
    for (int r = 0; r < 8; ++r) {
      int m   = r + hi * 8;
      int row = qbase + wave * 16 + m;            // t
      float inv = rsum[r] > 0.f ? 1.0f / rsum[r] : 0.f;
      Y[((size_t)b * TT + row) * TC + h * 64 + nt * 16 + ln] = f2b(o[nt][r] * inv);
    }
  }
}

// ---------------------------------------------------------------------------
// Launch
// ---------------------------------------------------------------------------
extern "C" void kernel_launch(void* const* d_in, const int* in_sizes, int n_in,
                              void* d_out, int out_size, void* d_ws, size_t ws_size,
                              hipStream_t stream) {
  const float* x      = (const float*)d_in[0];   // [B,T,C]
  const float* W_attn = (const float*)d_in[1];   // [3C,C]
  const float* b_attn = (const float*)d_in[2];   // [3C]
  const float* W_proj = (const float*)d_in[3];   // [C,C]
  const float* b_proj = (const float*)d_in[4];   // [C]
  float* out = (float*)d_out;                    // [B,T,C]

  const size_t M   = (size_t)TB * TT;            // 8192
  const size_t nX  = M * TC;                     // 8,388,608
  const size_t nWA = (size_t)3 * TC * TC;        // 3,145,728
  const size_t nWP = (size_t)TC * TC;            // 1,048,576
  const size_t per = (size_t)TB * TH * TT * TE;  // 8,388,608 (= nX)

  char* ws = (char*)d_ws;
  bf16* xbf  = (bf16*)(ws);                                   // 16 MiB
  bf16* wabf = (bf16*)(ws + nX * 2);                          //  6 MiB
  bf16* wpbf = (bf16*)(ws + nX * 2 + nWA * 2);                //  2 MiB
  bf16* qkv  = (bf16*)(ws + nX * 2 + nWA * 2 + nWP * 2);      // 48 MiB
  bf16* ybf  = (bf16*)(ws + nX * 2 + nWA * 2 + nWP * 2 + 3 * per * 2); // 16 MiB

  // 1) fp32 -> bf16 conversions
  f2bf_kernel<<<4096, 256, 0, stream>>>(x, xbf, nX);
  f2bf_kernel<<<2048, 256, 0, stream>>>(W_attn, wabf, nWA);
  f2bf_kernel<<<1024, 256, 0, stream>>>(W_proj, wpbf, nWP);

  // 2) QKV projection: [8192,1024] @ [3072,1024]^T, scatter to q/k/v heads
  gemm_xwT<<<dim3(3 * TC / 128, M / 128), 256, 0, stream>>>(
      xbf, wabf, b_attn, (int)M, 3 * TC, TC, /*mode=*/1, nullptr, qkv);

  // 3) Flash attention per (b,h,q-tile)
  attn_kernel<<<dim3(TT / 64, TB * TH), 128, 0, stream>>>(
      qkv, qkv + per, qkv + 2 * per, ybf);

  // 4) Output projection: [8192,1024] @ [1024,1024]^T + b -> fp32 out
  gemm_xwT<<<dim3(TC / 128, M / 128), 256, 0, stream>>>(
      ybf, wpbf, b_proj, (int)M, TC, TC, /*mode=*/0, out, nullptr);
}
